// FixedVectorQuantizerGCN_87041807220996
// MI455X (gfx1250) — matrix operand order repaired and checked by reading, not compiled, not run on hardware
//
#include <hip/hip_runtime.h>

// ---------------------------------------------------------------------------
// FixedVectorQuantizerGCN for MI455X (gfx1250, wave32, WMMA)
//
// All GEMMs run through v_wmma_f32_16x16x32_bf16 with a bf16x3 decomposition
// (hi/lo split of both operands, 3 WMMA products, fp32 accumulation) which
// recovers ~fp32 accuracy -- important because the argmin is a hard decision.
// ---------------------------------------------------------------------------

typedef __attribute__((ext_vector_type(16))) __bf16 v16bf;
typedef __attribute__((ext_vector_type(8)))  __bf16 v8bf;
typedef __attribute__((ext_vector_type(4)))  __bf16 v4bf;
typedef __attribute__((ext_vector_type(8)))  float  v8f;

#define TILE_M 128
#define TILE_N 128
#define TILE_K 32

struct bf2 { __bf16 hi; __bf16 lo; };

__device__ __forceinline__ bf2 split2(float f) {
  bf2 r;
  r.hi = (__bf16)f;
  r.lo = (__bf16)(f - (float)r.hi);
  return r;
}

// MODE 0: C = A@B
// MODE 1: C = relu(C + A@B + bias[col])          (GCN combine pass 2)
// MODE 2: C = 2*(A@B) - rowterm[row] - colterm[col]   (-distance epilogue)
// TRANSB 0: B is [Kc,N] row-major ("NN");  TRANSB 1: B is [N,Kc] row-major ("NT")
template <int TRANSB, int MODE>
__global__ __launch_bounds__(256)
void gemm_bf16x3(const float* __restrict__ A, const float* __restrict__ Bm,
                 float* __restrict__ C, int M, int N, int Kc,
                 const float* __restrict__ bias,
                 const float* __restrict__ rowterm,
                 const float* __restrict__ colterm)
{
  __shared__ __align__(32) __bf16 Ah[TILE_M][TILE_K];
  __shared__ __align__(32) __bf16 Al[TILE_M][TILE_K];
  __shared__ __align__(32) __bf16 Bh[TILE_N][TILE_K];   // stored [n][k]
  __shared__ __align__(32) __bf16 Bl[TILE_N][TILE_K];

  const int tid  = threadIdx.x;
  const int wave = tid >> 5;
  const int lane = tid & 31;
  const int hh   = lane >> 4;     // lane half (0/1)
  const int l16  = lane & 15;
  const int m0 = blockIdx.x * TILE_M;
  const int n0 = blockIdx.y * TILE_N;
  const int wm = (wave >> 2) * 64;   // wave M offset inside block (0/64)
  const int wn = (wave & 3) * 32;    // wave N offset inside block (0..96)

  v8f acc[4][2];
  const v8f vzero = {0.f,0.f,0.f,0.f,0.f,0.f,0.f,0.f};
#pragma unroll
  for (int mi = 0; mi < 4; ++mi)
#pragma unroll
    for (int ni = 0; ni < 2; ++ni) acc[mi][ni] = vzero;

  for (int k0 = 0; k0 < Kc; k0 += TILE_K) {
    // ---- stage A tile [128 x 32] as hi/lo bf16 into LDS ----
#pragma unroll
    for (int i = 0; i < 4; ++i) {
      int idx = tid + i * 256;           // 0..1023
      int r   = idx >> 3;                // 0..127
      int cg  = (idx & 7) << 2;          // 0,4,..,28
      const float4 v = *(const float4*)(A + (size_t)(m0 + r) * Kc + k0 + cg);
      v4bf hi, lo;
      bf2 s0 = split2(v.x); hi[0] = s0.hi; lo[0] = s0.lo;
      bf2 s1 = split2(v.y); hi[1] = s1.hi; lo[1] = s1.lo;
      bf2 s2 = split2(v.z); hi[2] = s2.hi; lo[2] = s2.lo;
      bf2 s3 = split2(v.w); hi[3] = s3.hi; lo[3] = s3.lo;
      *(v4bf*)&Ah[r][cg] = hi;
      *(v4bf*)&Al[r][cg] = lo;
    }
    // ---- stage B tile as [n][k] hi/lo bf16 into LDS ----
    if constexpr (TRANSB) {
#pragma unroll
      for (int i = 0; i < 4; ++i) {
        int idx = tid + i * 256;
        int r   = idx >> 3;              // n index 0..127
        int cg  = (idx & 7) << 2;        // k group
        const float4 v = *(const float4*)(Bm + (size_t)(n0 + r) * Kc + k0 + cg);
        v4bf hi, lo;
        bf2 s0 = split2(v.x); hi[0] = s0.hi; lo[0] = s0.lo;
        bf2 s1 = split2(v.y); hi[1] = s1.hi; lo[1] = s1.lo;
        bf2 s2 = split2(v.z); hi[2] = s2.hi; lo[2] = s2.lo;
        bf2 s3 = split2(v.w); hi[3] = s3.hi; lo[3] = s3.lo;
        *(v4bf*)&Bh[r][cg] = hi;
        *(v4bf*)&Bl[r][cg] = lo;
      }
    } else {
#pragma unroll
      for (int i = 0; i < 4; ++i) {
        int idx = tid + i * 256;
        int kr  = idx >> 5;              // 0..31
        int ncg = (idx & 31) << 2;       // 0..124
        const float4 v = *(const float4*)(Bm + (size_t)(k0 + kr) * N + n0 + ncg);
        bf2 s0 = split2(v.x); Bh[ncg + 0][kr] = s0.hi; Bl[ncg + 0][kr] = s0.lo;
        bf2 s1 = split2(v.y); Bh[ncg + 1][kr] = s1.hi; Bl[ncg + 1][kr] = s1.lo;
        bf2 s2 = split2(v.z); Bh[ncg + 2][kr] = s2.hi; Bl[ncg + 2][kr] = s2.lo;
        bf2 s3 = split2(v.w); Bh[ncg + 3][kr] = s3.hi; Bl[ncg + 3][kr] = s3.lo;
      }
    }
    // prefetch next k-tile while we compute this one (global_prefetch_b8)
    if (k0 + TILE_K < Kc) {
      __builtin_prefetch(A + (size_t)(m0 + (tid >> 1)) * Kc + k0 + TILE_K, 0, 0);
      if constexpr (TRANSB)
        __builtin_prefetch(Bm + (size_t)(n0 + (tid >> 1)) * Kc + k0 + TILE_K, 0, 0);
      else
        __builtin_prefetch(Bm + (size_t)(k0 + TILE_K + (tid >> 3)) * N + n0, 0, 0);
    }
    __syncthreads();

    // ---- build fragments in contiguous k order (same for A and B) ----
    // A (16-bit 16x32): lane holds row M=l16, K = hh*16 + e (matches B)
    v16bf ah[4], al[4];
#pragma unroll
    for (int mi = 0; mi < 4; ++mi) {
      int m = wm + mi * 16 + l16;
      v8bf p0 = *(const v8bf*)&Ah[m][hh * 16];
      v8bf p1 = *(const v8bf*)&Ah[m][hh * 16 + 8];
      ah[mi] = __builtin_shufflevector(p0, p1, 0,1,2,3,4,5,6,7,8,9,10,11,12,13,14,15);
      v8bf q0 = *(const v8bf*)&Al[m][hh * 16];
      v8bf q1 = *(const v8bf*)&Al[m][hh * 16 + 8];
      al[mi] = __builtin_shufflevector(q0, q1, 0,1,2,3,4,5,6,7,8,9,10,11,12,13,14,15);
    }
    // B: lane holds column N=l16, K = hh*16 + e (contiguous 32B in LDS)
#pragma unroll
    for (int ni = 0; ni < 2; ++ni) {
      int n = wn + ni * 16 + l16;
      v16bf bh = *(const v16bf*)&Bh[n][hh * 16];
      v16bf bl = *(const v16bf*)&Bl[n][hh * 16];
#pragma unroll
      for (int mi = 0; mi < 4; ++mi) {
        acc[mi][ni] = __builtin_amdgcn_wmma_f32_16x16x32_bf16(
            false, ah[mi], false, bh, (short)0, acc[mi][ni], false, false);
        acc[mi][ni] = __builtin_amdgcn_wmma_f32_16x16x32_bf16(
            false, ah[mi], false, bl, (short)0, acc[mi][ni], false, false);
        acc[mi][ni] = __builtin_amdgcn_wmma_f32_16x16x32_bf16(
            false, al[mi], false, bh, (short)0, acc[mi][ni], false, false);
      }
    }
    __syncthreads();
  }

  // ---- epilogue: C/D layout: element r -> M = hh*8 + r, N = l16 ----
#pragma unroll
  for (int mi = 0; mi < 4; ++mi) {
#pragma unroll
    for (int ni = 0; ni < 2; ++ni) {
      int col = n0 + wn + ni * 16 + l16;
#pragma unroll
      for (int r = 0; r < 8; ++r) {
        int row = m0 + wm + mi * 16 + hh * 8 + r;
        size_t off = (size_t)row * N + col;
        float v = acc[mi][ni][r];
        if constexpr (MODE == 0) {
          C[off] = v;
        } else if constexpr (MODE == 1) {
          float s = v + C[off] + bias[col];
          C[off] = s > 0.f ? s : 0.f;
        } else {
          C[off] = 2.f * v - rowterm[row] - colterm[col];
        }
      }
    }
  }
}

// ---------------------------------------------------------------------------
// per-row squared L2 norm (one block per row, 1024 cols)
__global__ __launch_bounds__(256)
void rownorm_kernel(const float* __restrict__ X, float* __restrict__ out, int ncols)
{
  __shared__ float red[256];
  const int row = blockIdx.x;
  const int t = threadIdx.x;
  const float* p = X + (size_t)row * ncols;
  float s = 0.f;
  for (int e = t * 4; e < ncols; e += 1024) {
    float4 v = *(const float4*)(p + e);
    s += v.x * v.x + v.y * v.y + v.z * v.z + v.w * v.w;
  }
  red[t] = s;
  __syncthreads();
  for (int off = 128; off > 0; off >>= 1) {
    if (t < off) red[t] += red[t + off];
    __syncthreads();
  }
  if (t == 0) out[row] = red[0];
}

// ---------------------------------------------------------------------------
// argmax over a row of -distance (== argmin distance, first-index tie-break),
// then gather the winning codebook row into quantized output.
__global__ __launch_bounds__(256)
void argmax_gather_kernel(const float* __restrict__ negd, const float* __restrict__ LM,
                          float* __restrict__ qout, int K, int E)
{
  __shared__ float sv[256];
  __shared__ int   si[256];
  const int b = blockIdx.x;
  const int t = threadIdx.x;
  const float* row = negd + (size_t)b * K;
  float best = -3.402823466e38f;
  int bi = 0x7fffffff;
  for (int k = t; k < K; k += 256) {   // ascending -> strict '>' keeps first max
    float v = row[k];
    if (v > best) { best = v; bi = k; }
  }
  sv[t] = best; si[t] = bi;
  __syncthreads();
  for (int off = 128; off > 0; off >>= 1) {
    if (t < off) {
      float ov = sv[t + off]; int oi = si[t + off];
      if (ov > sv[t] || (ov == sv[t] && oi < si[t])) { sv[t] = ov; si[t] = oi; }
    }
    __syncthreads();
  }
  const int idx = si[0];
  const float* src = LM + (size_t)idx * E;
  float* dst = qout + (size_t)b * E;
  for (int e = t * 4; e < E; e += 1024) {
    *(float4*)(dst + e) = *(const float4*)(src + e);
  }
}

// ---------------------------------------------------------------------------
extern "C" void kernel_launch(void* const* d_in, const int* in_sizes, int n_in,
                              void* d_out, int out_size, void* d_ws, size_t ws_size,
                              hipStream_t stream)
{
  constexpr int Bn = 16384, K = 1024, E = 1024;
  const float* x    = (const float*)d_in[0];
  // d_in[1] = var, unused by the reference
  const float* lm0  = (const float*)d_in[2];
  const float* adjp = (const float*)d_in[3];
  const float* adjc = (const float*)d_in[4];
  const float* Wp1  = (const float*)d_in[5];
  const float* Wc1  = (const float*)d_in[6];
  const float* b1   = (const float*)d_in[7];
  const float* Wp2  = (const float*)d_in[8];
  const float* Wc2  = (const float*)d_in[9];
  const float* b2   = (const float*)d_in[10];

  float* qout = (float*)d_out;                 // [B,E] quantized (== q forward)
  float* negd = qout + (size_t)Bn * E;         // [B,K] -distance

  // GCN intermediates live inside the (not-yet-final) -distance region;
  // everything here is dead before the distance GEMM rewrites it.
  const size_t KE = (size_t)K * E;             // 1M floats = 4MB
  float* T1 = negd;
  float* T2 = negd + KE;
  float* Hh = negd + 2 * KE;
  float* T3 = negd + 3 * KE;
  float* T4 = negd + 4 * KE;

  // LM must survive the distance GEMM + gather -> workspace (~4.1 MB)
  float* LM  = (float*)d_ws;
  float* lmn = LM + KE;                        // [K]
  float* xn  = lmn + K;                        // [B]

  dim3 blk(256);
  dim3 gS(K / TILE_M, E / TILE_N);             // 8x8 for the 1024^3 GEMMs

  // ---- GCN layer 1 ----
  gemm_bf16x3<0,0><<<gS, blk, 0, stream>>>(lm0,  Wp1, T1, K, E, E, nullptr, nullptr, nullptr);
  gemm_bf16x3<0,0><<<gS, blk, 0, stream>>>(lm0,  Wc1, T2, K, E, E, nullptr, nullptr, nullptr);
  gemm_bf16x3<0,0><<<gS, blk, 0, stream>>>(adjp, T1,  Hh, K, E, K, nullptr, nullptr, nullptr);
  gemm_bf16x3<0,1><<<gS, blk, 0, stream>>>(adjc, T2,  Hh, K, E, K, b1,      nullptr, nullptr);
  // ---- GCN layer 2 ----
  gemm_bf16x3<0,0><<<gS, blk, 0, stream>>>(Hh,   Wp2, T3, K, E, E, nullptr, nullptr, nullptr);
  gemm_bf16x3<0,0><<<gS, blk, 0, stream>>>(Hh,   Wc2, T4, K, E, E, nullptr, nullptr, nullptr);
  gemm_bf16x3<0,0><<<gS, blk, 0, stream>>>(adjp, T3,  LM, K, E, K, nullptr, nullptr, nullptr);
  gemm_bf16x3<0,1><<<gS, blk, 0, stream>>>(adjc, T4,  LM, K, E, K, b2,      nullptr, nullptr);

  // ---- norms ----
  rownorm_kernel<<<dim3(K),  blk, 0, stream>>>(LM, lmn, E);
  rownorm_kernel<<<dim3(Bn), blk, 0, stream>>>(x,  xn,  E);

  // ---- -distance = 2*x@LM^T - |x|^2 - |lm|^2  (overwrites the whole region) ----
  gemm_bf16x3<1,2><<<dim3(Bn / TILE_M, K / TILE_N), blk, 0, stream>>>(
      x, LM, negd, Bn, K, E, nullptr, xn, lmn);

  // ---- argmin + codebook gather ----
  argmax_gather_kernel<<<dim3(Bn), blk, 0, stream>>>(negd, LM, qout, K, E);

  (void)in_sizes; (void)n_in; (void)out_size; (void)ws_size;
}
